// GNNAutoencoder_21990232555807
// MI455X (gfx1250) — matrix operand order repaired and checked by reading, not compiled, not run on hardware
//
#include <hip/hip_runtime.h>
#include <hip/hip_bf16.h>

typedef __attribute__((ext_vector_type(16))) _Float16 v16h;
typedef __attribute__((ext_vector_type(8)))  _Float16 v8h;
typedef __attribute__((ext_vector_type(8)))  float    v8f;

namespace gnn {

constexpr int N_NODES = 50000;
constexpr int N_EDGES = 800000;
constexpr int N_EP    = 850000;      // edges + self loops
constexpr int F_IN    = 128;
constexpr int HEADS   = 4;
constexpr int HID     = 64;
constexpr int H1C     = 256;         // HEADS*HID
constexpr int LAT     = 64;
constexpr int N_EC    = 8;
constexpr float SLOPE = 0.2f;

// ---------- CDNA5 async global->LDS copy (ASYNCcnt) ----------
__device__ __forceinline__ void async_copy_b128(unsigned lds_off, const void* gaddr) {
    asm volatile("global_load_async_to_lds_b128 %0, %1, off"
                 :: "v"(lds_off), "v"(gaddr) : "memory");
}
__device__ __forceinline__ void wait_async() {
    asm volatile("s_wait_asynccnt 0" ::: "memory");
}

// ---------- float <-> orderable uint (for atomic max on floats) ----------
__device__ __forceinline__ unsigned fmap(float f) {
    unsigned u = __float_as_uint(f);
    return (u & 0x80000000u) ? ~u : (u | 0x80000000u);
}
__device__ __forceinline__ float funmap(unsigned u) {
    return (u & 0x80000000u) ? __uint_as_float(u ^ 0x80000000u)
                             : __uint_as_float(~u);
}
constexpr unsigned NEG_INF_MAPPED = 0x007FFFFFu;   // fmap(-inf)

// ---------- WMMA wave32 B-fragment index map (ISA 7.12.2, 16-bit B 32x16) ----------
__device__ __forceinline__ int b_frag_k(int lane, int i) {
    return ((i & 8) ? 16 : 0) + ((lane & 16) ? 8 : 0) + (i & 7);
}

// Build a lane's A fragment from an LDS row (two contiguous 8-half chunks).
__device__ __forceinline__ v16h load_a_frag(const _Float16* rowBase, int k0, int lane) {
    const int klo = k0 + ((lane & 16) ? 8 : 0);
    v8h lo = *(const v8h*)(rowBase + klo);
    v8h hi = *(const v8h*)(rowBase + klo + 16);
    return __builtin_shufflevector(lo, hi, 0, 1, 2, 3, 4, 5, 6, 7,
                                           8, 9, 10, 11, 12, 13, 14, 15);
}

// ---------- small utility kernels ----------
__global__ void k_f32_to_f16(const float* __restrict__ in, _Float16* __restrict__ out, int n) {
    int t = blockIdx.x * blockDim.x + threadIdx.x;
    if (t < n) out[t] = (_Float16)in[t];
}

__global__ void k_fill_u32(unsigned* __restrict__ p, unsigned v, int n) {
    int t = blockIdx.x * blockDim.x + threadIdx.x;
    if (t < n) p[t] = v;
}

__global__ void k_bias_relu_f16(const float* __restrict__ agg, const float* __restrict__ bias,
                                _Float16* __restrict__ out, int ncols, int total) {
    int t = blockIdx.x * blockDim.x + threadIdx.x;
    if (t >= total) return;
    float v = agg[t] + bias[t % ncols];
    out[t] = (_Float16)fmaxf(v, 0.0f);
}

__global__ void k_bias_f16(const float* __restrict__ agg, const float* __restrict__ bias,
                           _Float16* __restrict__ out, int ncols, int total) {
    int t = blockIdx.x * blockDim.x + threadIdx.x;
    if (t >= total) return;
    out[t] = (_Float16)(agg[t] + bias[t % ncols]);
}

// ---------- pre-swizzle f32 weights [KD x ND] into WMMA B-fragment order (f16) ----------
template<int KD, int ND>
__global__ void k_pack_b(const float* __restrict__ B, _Float16* __restrict__ Bsw) {
    int t = blockIdx.x * blockDim.x + threadIdx.x;
    if (t >= KD * ND) return;
    int i    = t & 15;
    int lane = (t >> 4) & 31;
    int f    = t >> 9;               // fragment id = nt*(KD/32) + ks
    int ks   = f % (KD / 32);
    int nt   = f / (KD / 32);
    int k    = ks * 32 + b_frag_k(lane, i);
    int n    = nt * 16 + (lane & 15);
    Bsw[t] = (_Float16)B[(size_t)k * ND + n];
}

// ---------- generic tiled WMMA GEMM: out[M x ND] = A[M x KD] * Bsw ----------
template<int KD, int ND, int NWAVES, bool OUT16, bool BIAS>
__global__ void k_gemm(const _Float16* __restrict__ A, const _Float16* __restrict__ Bsw,
                       const float* __restrict__ bias,
                       _Float16* __restrict__ out16, float* __restrict__ out32) {
    constexpr int KSTEPS = KD / 32;
    __shared__ _Float16 lA[16 * KD];
    const int m0  = blockIdx.x * 16;
    const int tid = threadIdx.x;

    // A tile = 16 full contiguous rows -> async DMA straight into LDS (no VGPR round trip)
    {
        const char* srcA   = (const char*)(A + (size_t)m0 * KD);
        unsigned    ldsA   = (unsigned)(size_t)lA;
        constexpr int CHUNKS = 16 * KD * 2 / 16;           // 16-byte chunks
        for (int i = tid; i < CHUNKS; i += NWAVES * 32)
            async_copy_b128(ldsA + i * 16, srcA + (size_t)i * 16);
        wait_async();
    }
    __syncthreads();

    const int lane  = tid & 31;
    const int wave  = tid >> 5;
    const int col   = lane & 15;
    const int rbase = (lane & 16) ? 8 : 0;
    const _Float16* rowBase = lA + col * KD;

    for (int nt = wave; nt < ND / 16; nt += NWAVES) {
        const int n0 = nt * 16;
        const _Float16* bp = Bsw + ((size_t)nt * KSTEPS * 32 + lane) * 16;
        v8f c = {0.f, 0.f, 0.f, 0.f, 0.f, 0.f, 0.f, 0.f};
#pragma unroll
        for (int ks = 0; ks < KSTEPS; ++ks) {
            v16h a = load_a_frag(rowBase, ks * 32, lane);
            v16h b = *(const v16h*)(bp + (size_t)ks * 32 * 16);
            c = __builtin_amdgcn_wmma_f32_16x16x32_f16(false, a, false, b,
                                                       (short)0, c, false, false);
        }
#pragma unroll
        for (int r = 0; r < 8; ++r) {
            const int m = m0 + rbase + r;
            const int n = n0 + col;
            float v = c[r];
            if (BIAS) v += bias[n];
            if (OUT16) out16[(size_t)m * ND + n] = (_Float16)v;
            else       out32[(size_t)m * ND + n] = v;
        }
    }
}

// ---------- attention score dots: a_s[n,h] = <h[n,h,:], att_src[h,:]> ----------
template<int H, int C>
__global__ void k_att_scores(const _Float16* __restrict__ hfeat,
                             const float* __restrict__ att_s, const float* __restrict__ att_d,
                             float* __restrict__ as_, float* __restrict__ ad_) {
    int t = blockIdx.x * blockDim.x + threadIdx.x;
    if (t >= N_NODES * H) return;
    int n = t / H, h = t % H;
    const v8h* p8 = (const v8h*)(hfeat + (size_t)n * H * C + h * C);
    float as = 0.f, ad = 0.f;
#pragma unroll
    for (int c8 = 0; c8 < C / 8; ++c8) {
        v8h v = p8[c8];
#pragma unroll
        for (int j = 0; j < 8; ++j) {
            float vf = (float)v[j];
            as += vf * att_s[h * C + c8 * 8 + j];
            ad += vf * att_d[h * C + c8 * 8 + j];
        }
    }
    as_[t] = as;
    ad_[t] = ad;
}

__device__ __forceinline__ void edge_sd(const int* __restrict__ ei, int e, int& s, int& d) {
    if (e < N_EDGES) { s = ei[e]; d = ei[N_EDGES + e]; }
    else             { s = e - N_EDGES; d = s; }          // appended self loops
}

template<int H>
__global__ void k_edge_max(const int* __restrict__ ei, const float* __restrict__ as_,
                           const float* __restrict__ ad_, unsigned* __restrict__ amax) {
    int t = blockIdx.x * blockDim.x + threadIdx.x;
    if (t >= N_EP * H) return;
    int e = t / H, h = t % H, s, d;
    edge_sd(ei, e, s, d);
    float al = as_[s * H + h] + ad_[d * H + h];
    al = (al > 0.f) ? al : SLOPE * al;
    atomicMax(&amax[d * H + h], fmap(al));
}

template<int H>
__global__ void k_edge_exp(const int* __restrict__ ei, const float* __restrict__ as_,
                           const float* __restrict__ ad_, const unsigned* __restrict__ amax,
                           float* __restrict__ exbuf, float* __restrict__ denom) {
    int t = blockIdx.x * blockDim.x + threadIdx.x;
    if (t >= N_EP * H) return;
    int e = t / H, h = t % H, s, d;
    edge_sd(ei, e, s, d);
    float al = as_[s * H + h] + ad_[d * H + h];
    al = (al > 0.f) ? al : SLOPE * al;
    float ex = __expf(al - funmap(amax[d * H + h]));
    exbuf[t] = ex;
    atomicAdd(&denom[d * H + h], ex);
}

// one wave (32 lanes) per edge: agg[dst] += h[src] * alpha
template<int H, int C>
__global__ void k_edge_agg(const int* __restrict__ ei, const _Float16* __restrict__ hfeat,
                           const float* __restrict__ exbuf, const float* __restrict__ denom,
                           float* __restrict__ agg) {
    int wid  = (int)((blockIdx.x * blockDim.x + threadIdx.x) >> 5);
    int lane = threadIdx.x & 31;
    if (wid >= N_EP) return;
    int s, d;
    edge_sd(ei, wid, s, d);
    float w[H];
#pragma unroll
    for (int h = 0; h < H; ++h)
        w[h] = exbuf[(size_t)wid * H + h] / (denom[d * H + h] + 1e-16f);
    const _Float16* ps = hfeat + (size_t)s * H * C;
    float* pd = agg + (size_t)d * H * C;
    for (int idx = lane; idx < H * C; idx += 32)
        atomicAdd(&pd[idx], (float)ps[idx] * w[idx / C]);
}

// ---------- edge decoder: 16 edges per block, async gather + WMMA + tiny 64x8 layer ----------
__global__ void k_edge_decode(const int* __restrict__ ei, const _Float16* __restrict__ zh,
                              const _Float16* __restrict__ We1sw, const float* __restrict__ be1,
                              const float* __restrict__ We2, const float* __restrict__ be2,
                              float* __restrict__ out) {
    __shared__ _Float16 lA[16 * 128];   // concat(z[src], z[dst]) for 16 edges
    __shared__ float    T[16 * 64];     // relu(edge_z @ We1 + be1)
    __shared__ float    We2s[64 * N_EC];
    const int e0  = blockIdx.x * 16;
    const int tid = threadIdx.x;        // 128 threads = 4 waves

    // gather via async DMA: 256 chunks of 8 halfs (16B), 16 chunks per edge
    {
        unsigned ldsA = (unsigned)(size_t)lA;
        for (int cc = tid; cc < 256; cc += 128) {
            int e  = cc >> 4;
            int c0 = (cc & 15) * 8;                        // starting half within 128
            int node = (c0 < 64) ? ei[e0 + e] : ei[N_EDGES + e0 + e];
            async_copy_b128(ldsA + cc * 16, zh + (size_t)node * LAT + (c0 & 63));
        }
    }
    for (int i = tid; i < 64 * N_EC; i += 128) We2s[i] = We2[i];
    wait_async();
    __syncthreads();

    const int lane  = tid & 31;
    const int wave  = tid >> 5;
    const int col   = lane & 15;
    const int rbase = (lane & 16) ? 8 : 0;
    const int n0    = wave * 16;        // 4 waves cover ND = 64
    const _Float16* rowBase = lA + col * 128;
    const _Float16* bp = We1sw + ((size_t)wave * 4 * 32 + lane) * 16;   // KSTEPS=4

    v8f c = {0.f, 0.f, 0.f, 0.f, 0.f, 0.f, 0.f, 0.f};
#pragma unroll
    for (int ks = 0; ks < 4; ++ks) {
        v16h a = load_a_frag(rowBase, ks * 32, lane);
        v16h b = *(const v16h*)(bp + (size_t)ks * 32 * 16);
        c = __builtin_amdgcn_wmma_f32_16x16x32_f16(false, a, false, b,
                                                   (short)0, c, false, false);
    }
#pragma unroll
    for (int r = 0; r < 8; ++r)
        T[(rbase + r) * 64 + n0 + col] = fmaxf(c[r] + be1[n0 + col], 0.0f);
    __syncthreads();

    // second layer: 16 edges x 8 classes = 128 outputs, all operands in LDS
    int e  = tid >> 3;
    int oc = tid & 7;
    float acc = be2[oc];
#pragma unroll
    for (int k = 0; k < 64; ++k)
        acc += T[e * 64 + k] * We2s[k * N_EC + oc];
    out[(size_t)(e0 + e) * N_EC + oc] = acc;
}

} // namespace gnn

extern "C" void kernel_launch(void* const* d_in, const int* in_sizes, int n_in,
                              void* d_out, int out_size, void* d_ws, size_t ws_size,
                              hipStream_t stream) {
    using namespace gnn;
    (void)in_sizes; (void)n_in; (void)out_size; (void)ws_size;

    const float* x    = (const float*)d_in[0];
    const int*   ei   = (const int*)d_in[1];
    const float* W1   = (const float*)d_in[2];
    const float* as1v = (const float*)d_in[3];
    const float* ad1v = (const float*)d_in[4];
    const float* b1   = (const float*)d_in[5];
    const float* W2   = (const float*)d_in[6];
    const float* as2v = (const float*)d_in[7];
    const float* ad2v = (const float*)d_in[8];
    const float* b2   = (const float*)d_in[9];
    const float* Wn   = (const float*)d_in[10];
    const float* bn   = (const float*)d_in[11];
    const float* We1  = (const float*)d_in[12];
    const float* be1  = (const float*)d_in[13];
    const float* We2  = (const float*)d_in[14];
    const float* be2  = (const float*)d_in[15];

    char* wsp = (char*)d_ws;
    size_t off = 0;
    auto alloc = [&](size_t bytes) -> void* {
        void* p = wsp + off;
        off += (bytes + 255) & ~(size_t)255;
        return p;
    };

    _Float16* xh    = (_Float16*)alloc((size_t)N_NODES * F_IN * 2);
    _Float16* W1sw  = (_Float16*)alloc((size_t)F_IN * H1C * 2);
    _Float16* W2sw  = (_Float16*)alloc((size_t)H1C * LAT * 2);
    _Float16* Wnsw  = (_Float16*)alloc((size_t)LAT * F_IN * 2);
    _Float16* We1sw = (_Float16*)alloc((size_t)2 * LAT * HID * 2);
    _Float16* h1h   = (_Float16*)alloc((size_t)N_NODES * H1C * 2);
    float*    as1   = (float*)alloc((size_t)N_NODES * HEADS * 4);
    float*    ad1   = (float*)alloc((size_t)N_NODES * HEADS * 4);
    unsigned* amax1 = (unsigned*)alloc((size_t)N_NODES * HEADS * 4);
    float*    den1  = (float*)alloc((size_t)N_NODES * HEADS * 4);
    float*    ex1   = (float*)alloc((size_t)N_EP * HEADS * 4);
    float*    agg1  = (float*)alloc((size_t)N_NODES * H1C * 4);
    _Float16* hh    = (_Float16*)alloc((size_t)N_NODES * H1C * 2);
    _Float16* t2h   = (_Float16*)alloc((size_t)N_NODES * LAT * 2);
    float*    as2   = (float*)alloc((size_t)N_NODES * 4);
    float*    ad2   = (float*)alloc((size_t)N_NODES * 4);
    unsigned* amax2 = (unsigned*)alloc((size_t)N_NODES * 4);
    float*    den2  = (float*)alloc((size_t)N_NODES * 4);
    float*    ex2   = (float*)alloc((size_t)N_EP * 4);
    float*    agg2  = (float*)alloc((size_t)N_NODES * LAT * 4);
    _Float16* zh    = (_Float16*)alloc((size_t)N_NODES * LAT * 2);

    float* out_nodes = (float*)d_out;
    float* out_edges = out_nodes + (size_t)N_NODES * F_IN;

    auto cdiv = [](long long a, long long b) { return (int)((a + b - 1) / b); };

    // ---- input convert + weight fragment pre-swizzles ----
    k_f32_to_f16<<<cdiv((long long)N_NODES * F_IN, 256), 256, 0, stream>>>(x, xh, N_NODES * F_IN);
    k_pack_b<F_IN, H1C><<<cdiv(F_IN * H1C, 256), 256, 0, stream>>>(W1, W1sw);
    k_pack_b<H1C, LAT><<<cdiv(H1C * LAT, 256), 256, 0, stream>>>(W2, W2sw);
    k_pack_b<LAT, F_IN><<<cdiv(LAT * F_IN, 256), 256, 0, stream>>>(Wn, Wnsw);
    k_pack_b<2 * LAT, HID><<<cdiv(2 * LAT * HID, 256), 256, 0, stream>>>(We1, We1sw);

    // ---- init reduction buffers ----
    k_fill_u32<<<cdiv(N_NODES * HEADS, 256), 256, 0, stream>>>(amax1, NEG_INF_MAPPED, N_NODES * HEADS);
    k_fill_u32<<<cdiv(N_NODES, 256), 256, 0, stream>>>(amax2, NEG_INF_MAPPED, N_NODES);
    hipMemsetAsync(den1, 0, (size_t)N_NODES * HEADS * 4, stream);
    hipMemsetAsync(agg1, 0, (size_t)N_NODES * H1C * 4, stream);
    hipMemsetAsync(den2, 0, (size_t)N_NODES * 4, stream);
    hipMemsetAsync(agg2, 0, (size_t)N_NODES * LAT * 4, stream);

    // ---- layer 1: h1 = x @ W1 (WMMA) ----
    k_gemm<F_IN, H1C, 8, true, false><<<N_NODES / 16, 256, 0, stream>>>(xh, W1sw, nullptr, h1h, nullptr);
    k_att_scores<HEADS, HID><<<cdiv(N_NODES * HEADS, 256), 256, 0, stream>>>(h1h, as1v, ad1v, as1, ad1);
    k_edge_max<HEADS><<<cdiv((long long)N_EP * HEADS, 256), 256, 0, stream>>>(ei, as1, ad1, amax1);
    k_edge_exp<HEADS><<<cdiv((long long)N_EP * HEADS, 256), 256, 0, stream>>>(ei, as1, ad1, amax1, ex1, den1);
    k_edge_agg<HEADS, HID><<<cdiv((long long)N_EP * 32, 256), 256, 0, stream>>>(ei, h1h, ex1, den1, agg1);
    k_bias_relu_f16<<<cdiv((long long)N_NODES * H1C, 256), 256, 0, stream>>>(agg1, b1, hh, H1C, N_NODES * H1C);

    // ---- layer 2: t2 = h @ W2 (WMMA) ----
    k_gemm<H1C, LAT, 4, true, false><<<N_NODES / 16, 128, 0, stream>>>(hh, W2sw, nullptr, t2h, nullptr);
    k_att_scores<1, LAT><<<cdiv(N_NODES, 256), 256, 0, stream>>>(t2h, as2v, ad2v, as2, ad2);
    k_edge_max<1><<<cdiv(N_EP, 256), 256, 0, stream>>>(ei, as2, ad2, amax2);
    k_edge_exp<1><<<cdiv(N_EP, 256), 256, 0, stream>>>(ei, as2, ad2, amax2, ex2, den2);
    k_edge_agg<1, LAT><<<cdiv((long long)N_EP * 32, 256), 256, 0, stream>>>(ei, t2h, ex2, den2, agg2);
    k_bias_f16<<<cdiv((long long)N_NODES * LAT, 256), 256, 0, stream>>>(agg2, b2, zh, LAT, N_NODES * LAT);

    // ---- node decoder: recon_x = z @ Wn + bn (WMMA, f32 out) ----
    k_gemm<LAT, F_IN, 8, false, true><<<N_NODES / 16, 256, 0, stream>>>(zh, Wnsw, bn, nullptr, out_nodes);

    // ---- edge decoder (async gather + WMMA + small dense layer) ----
    k_edge_decode<<<N_EDGES / 16, 128, 0, stream>>>(ei, zh, We1sw, be1, We2, be2, out_edges);
}
